// SelfReferentialAttention_52742198395037
// MI455X (gfx1250) — matrix operand order repaired and checked by reading, not compiled, not run on hardware
//
#include <hip/hip_runtime.h>
#include <hip/hip_bf16.h>

typedef __bf16 bf16;
typedef __attribute__((ext_vector_type(16))) __bf16 v16bf;
typedef __attribute__((ext_vector_type(8)))  float   v8f;

#define DIMX 3584
#define SEQ  2048
#define NHX  28
#define NKVX 4
#define HDX  128
#define KVD  512
#define MEMH 512

union FragU { v16bf v; uint4 u[2]; };
union U8s   { uint4 u; unsigned short s[8]; };

__device__ __forceinline__ v8f zero8() {
  v8f r;
#pragma unroll
  for (int i = 0; i < 8; i++) r[i] = 0.f;
  return r;
}

// A-fragment (16x32 bf16, MxK): lane<16 = row lane, K=0..15 contiguous;
// lanes 16..31 = same rows, K=16..31 (matches load_frag_b).
__device__ __forceinline__ v16bf load_frag_a(const bf16* base, int ld) {
  int lane = threadIdx.x & 31;
  int row  = lane & 15;
  int half = lane >> 4;
  FragU f;
  f.u[0] = *(const uint4*)(base + row * ld + 16 * half);
  f.u[1] = *(const uint4*)(base + row * ld + 16 * half + 8);
  return f.v;
}

// B-fragment (32x16 bf16, KxN): lane<16 = column lane holding K=0..15 contiguous;
// lanes 16..31 hold K=16..31.  Source is row-major [N][K] (i.e. W or K-tile rows).
__device__ __forceinline__ v16bf load_frag_b(const bf16* base, int ld) {
  int lane = threadIdx.x & 31;
  int n    = lane & 15;
  int half = lane >> 4;
  const bf16* p = base + n * ld + 16 * half;
  FragU f;
  f.u[0] = ((const uint4*)p)[0];
  f.u[1] = ((const uint4*)p)[1];
  return f.v;
}

__device__ __forceinline__ v8f wmma_bf16(v16bf a, v16bf b, v8f c) {
  return __builtin_amdgcn_wmma_f32_16x16x32_bf16(false, a, false, b, (short)0, c,
                                                 false, false);
}

__device__ __forceinline__ float sigmoidf_(float x) {
  return 1.f / (1.f + __expf(-x));
}

// ---------------------------------------------------------------- RMSNorm -> bf16
__global__ __launch_bounds__(256) void rmsnorm_bf16_kernel(
    const float* __restrict__ x, const float* __restrict__ w, bf16* __restrict__ xb) {
  int s = blockIdx.x;
  const float* row = x + (size_t)s * DIMX;
  bf16* orow = xb + (size_t)s * DIMX;
  float ss = 0.f;
  for (int i = threadIdx.x; i < DIMX; i += 256) { float v = row[i]; ss += v * v; }
#pragma unroll
  for (int off = 16; off > 0; off >>= 1) ss += __shfl_xor(ss, off, 32);
  __shared__ float red[8];
  int wave = threadIdx.x >> 5, lane = threadIdx.x & 31;
  if (lane == 0) red[wave] = ss;
  __syncthreads();
  float tot = 0.f;
#pragma unroll
  for (int i = 0; i < 8; i++) tot += red[i];
  float sc = rsqrtf(tot / (float)DIMX + 1e-6f);
  for (int i = threadIdx.x; i < DIMX; i += 256)
    orow[i] = (bf16)(row[i] * sc * w[i]);
}

// --------------------------------------------------------------- tiled WMMA GEMM
// out[M,N] = A[M,K](bf16) @ W[N,K]^T (fp32 weights converted to bf16 in LDS)
// Block tile 128x128, 8 waves, per-wave tile 32x64 (2 A-frags x 4 B-frags -> 8 WMMA).
// Double-buffered LDS: global prefetch of tile k+1 overlaps WMMA on tile k.
enum { EP_F32_BIAS = 0, EP_BF16_BIAS = 1, EP_BF16_SILU = 2, EP_F32_SIGMOID = 3, EP_F32 = 4 };

template <int MODE>
__global__ __launch_bounds__(256) void gemm_wmma_kernel(
    const bf16* __restrict__ A, const float* __restrict__ W,
    const float* __restrict__ bias, void* __restrict__ out, int K, int N) {
  __shared__ __align__(16) bf16 As[2][128][32];
  __shared__ __align__(16) bf16 Ws[2][128][32];
  const int m0 = blockIdx.y * 128, n0 = blockIdx.x * 128;
  const int t = threadIdx.x;
  const int wave = t >> 5, lane = t & 31;
  const int wm = wave & 3, wn = wave >> 2;  // 4 M-strips (32 rows) x 2 N-strips (64 cols)
  v8f acc[2][4];
#pragma unroll
  for (int i = 0; i < 2; i++)
#pragma unroll
    for (int j = 0; j < 4; j++) acc[i][j] = zero8();

  const int rT = t >> 1, kcT = (t & 1) * 16;  // staging coords: 128 rows x 32 cols
  const bf16*  aRow = A + (size_t)(m0 + rT) * K + kcT;
  const float* wRow = W + (size_t)(n0 + rT) * K + kcT;

  uint4  aReg[2];
  float4 wReg[4];

  // prologue: load K-tile 0 and stage into buffer 0
  aReg[0] = ((const uint4*)aRow)[0];
  aReg[1] = ((const uint4*)aRow)[1];
#pragma unroll
  for (int i = 0; i < 4; i++) wReg[i] = ((const float4*)wRow)[i];
  *(uint4*)&As[0][rT][kcT]     = aReg[0];
  *(uint4*)&As[0][rT][kcT + 8] = aReg[1];
  {
    bf16* d = &Ws[0][rT][kcT];
#pragma unroll
    for (int i = 0; i < 4; i++) {
      float4 f = wReg[i];
      d[4 * i + 0] = (bf16)f.x; d[4 * i + 1] = (bf16)f.y;
      d[4 * i + 2] = (bf16)f.z; d[4 * i + 3] = (bf16)f.w;
    }
  }
  __syncthreads();

  int buf = 0;
  for (int k0 = 0; k0 < K; k0 += 32) {
    const bool more = (k0 + 32) < K;
    if (more) {  // prefetch next K-tile into registers (overlaps WMMA below)
      const bf16*  ap = aRow + (k0 + 32);
      const float* wp = wRow + (k0 + 32);
      aReg[0] = ((const uint4*)ap)[0];
      aReg[1] = ((const uint4*)ap)[1];
#pragma unroll
      for (int i = 0; i < 4; i++) wReg[i] = ((const float4*)wp)[i];
    }

    v16bf a0 = load_frag_a(&As[buf][32 * wm][0], 32);
    v16bf a1 = load_frag_a(&As[buf][32 * wm + 16][0], 32);
#pragma unroll
    for (int tn = 0; tn < 4; tn++) {
      v16bf b = load_frag_b(&Ws[buf][64 * wn + 16 * tn][0], 32);
      acc[0][tn] = wmma_bf16(a0, b, acc[0][tn]);
      acc[1][tn] = wmma_bf16(a1, b, acc[1][tn]);
    }

    if (more) {  // stage prefetched tile into the other buffer
      int nb = buf ^ 1;
      *(uint4*)&As[nb][rT][kcT]     = aReg[0];
      *(uint4*)&As[nb][rT][kcT + 8] = aReg[1];
      bf16* d = &Ws[nb][rT][kcT];
#pragma unroll
      for (int i = 0; i < 4; i++) {
        float4 f = wReg[i];
        d[4 * i + 0] = (bf16)f.x; d[4 * i + 1] = (bf16)f.y;
        d[4 * i + 2] = (bf16)f.z; d[4 * i + 3] = (bf16)f.w;
      }
    }
    __syncthreads();
    buf ^= 1;
  }

  const int half = lane >> 4, cn = lane & 15;
#pragma unroll
  for (int tm = 0; tm < 2; tm++) {
#pragma unroll
    for (int tn = 0; tn < 4; tn++) {
#pragma unroll
      for (int r = 0; r < 8; r++) {
        int row = m0 + 32 * wm + 16 * tm + 8 * half + r;
        int col = n0 + 64 * wn + 16 * tn + cn;
        float v = acc[tm][tn][r];
        if constexpr (MODE == EP_F32_BIAS || MODE == EP_BF16_BIAS) v += bias[col];
        if constexpr (MODE == EP_BF16_SILU) v = v * sigmoidf_(v);
        if constexpr (MODE == EP_F32_SIGMOID) v = sigmoidf_(v);
        size_t idx = (size_t)row * N + col;
        if constexpr (MODE == EP_BF16_BIAS || MODE == EP_BF16_SILU)
          ((bf16*)out)[idx] = (bf16)v;
        else
          ((float*)out)[idx] = v;
      }
    }
  }
}

// ------------------------------------------------- gate combine: static + g * mem
__global__ __launch_bounds__(256) void combine_gate_kernel(
    const float* __restrict__ S, const float* __restrict__ G,
    const float* __restrict__ M, bf16* __restrict__ out, int n) {
  int i = blockIdx.x * 256 + threadIdx.x;
  if (i < n) out[i] = (bf16)(S[i] + G[i] * M[i]);
}

// ---------------------------------------------------------- lr / decay heads [S,4]
__global__ __launch_bounds__(256) void lrdecay_kernel(
    const bf16* __restrict__ xb, const float* __restrict__ lrs,
    const float* __restrict__ lrm, const float* __restrict__ dcs,
    const float* __restrict__ dcm, float* __restrict__ lr_out,
    float* __restrict__ dc_out) {
  int s = blockIdx.x;
  const bf16* xr = xb + (size_t)s * DIMX;
  float acc[16];
#pragma unroll
  for (int j = 0; j < 16; j++) acc[j] = 0.f;
  for (int i = threadIdx.x; i < DIMX; i += 256) {
    float xv = (float)xr[i];
#pragma unroll
    for (int j = 0; j < 4; j++) {
      acc[j]      += xv * lrs[j * DIMX + i];
      acc[4 + j]  += xv * lrm[j * DIMX + i];
      acc[8 + j]  += xv * dcs[j * DIMX + i];
      acc[12 + j] += xv * dcm[j * DIMX + i];
    }
  }
#pragma unroll
  for (int j = 0; j < 16; j++) {
#pragma unroll
    for (int off = 16; off > 0; off >>= 1) acc[j] += __shfl_xor(acc[j], off, 32);
  }
  __shared__ float red[8][16];
  int wave = threadIdx.x >> 5, lane = threadIdx.x & 31;
  if (lane == 0) {
#pragma unroll
    for (int j = 0; j < 16; j++) red[wave][j] = acc[j];
  }
  __syncthreads();
  if (threadIdx.x < 4) {
    int j = threadIdx.x;
    float a = 0.f, b = 0.f, c = 0.f, d = 0.f;
#pragma unroll
    for (int w = 0; w < 8; w++) {
      a += red[w][j]; b += red[w][4 + j]; c += red[w][8 + j]; d += red[w][12 + j];
    }
    lr_out[s * 4 + j] = sigmoidf_(a + b);
    dc_out[s * 4 + j] = sigmoidf_(c + d);
  }
}

// --------------------------------------------- flash attention (causal GQA, WMMA)
__global__ __launch_bounds__(128) void flash_wmma_kernel(
    const bf16* __restrict__ Q, const bf16* __restrict__ Kc,
    const bf16* __restrict__ Vc, bf16* __restrict__ O) {
  __shared__ __align__(16) bf16 Qs[64][128];
  __shared__ __align__(16) bf16 Ks[64][128];
  __shared__ __align__(16) bf16 Vt[128][64];      // V transposed: [d][key]
  __shared__ __align__(16) bf16 Ps[4][16][64];    // per-wave P strip
  const int h = blockIdx.x;
  const int q0 = blockIdx.y * 64;
  const int hkv = h / (NHX / NKVX);
  const int t = threadIdx.x, wave = t >> 5, lane = t & 31;
  const int half = lane >> 4, ln = lane & 15;
  const float scale = 0.08838834764831845f;  // 1/sqrt(128)

#pragma unroll
  for (int i = 0; i < 8; i++) {  // load Q tile 64x128
    int idx = t + 128 * i;
    int r = idx >> 4, c = (idx & 15) * 8;
    *(uint4*)&Qs[r][c] =
        *(const uint4*)(Q + (size_t)(q0 + r) * (NHX * HDX) + h * HDX + c);
  }

  float m_i[8], l_i[8];
  v8f acc[8];
#pragma unroll
  for (int r = 0; r < 8; r++) { m_i[r] = -1e30f; l_i[r] = 0.f; }
#pragma unroll
  for (int nt = 0; nt < 8; nt++) acc[nt] = zero8();

  const int ktiles = q0 / 64 + 1;  // causal
  for (int kt = 0; kt < ktiles; kt++) {
    int c0 = kt * 64;
    // stage K (row-major, vector) and V (transposed, row-pairs packed to b32)
#pragma unroll
    for (int i = 0; i < 4; i++) {
      int idx = t + 128 * i;                 // 0..511
      int pr = idx >> 4;                     // row pair 0..31
      int r  = pr * 2;
      int c  = (idx & 15) * 8;               // d-chunk
      const bf16* k0p = Kc + (size_t)(c0 + r) * KVD + hkv * HDX + c;
      *(uint4*)&Ks[r][c]     = ((const uint4*)k0p)[0];
      *(uint4*)&Ks[r + 1][c] = *(const uint4*)(k0p + KVD);
      const bf16* v0p = Vc + (size_t)(c0 + r) * KVD + hkv * HDX + c;
      U8s u0, u1;
      u0.u = ((const uint4*)v0p)[0];
      u1.u = *(const uint4*)(v0p + KVD);
#pragma unroll
      for (int j = 0; j < 8; j++) {
        unsigned int pk = (unsigned int)u0.s[j] | ((unsigned int)u1.s[j] << 16);
        *(unsigned int*)&Vt[c + j][r] = pk;  // keys r, r+1 of dim c+j
      }
    }
    __syncthreads();

    v8f sc[4];  // score strip 16x64 (wave rows 16*wave..+15)
#pragma unroll
    for (int tn = 0; tn < 4; tn++) sc[tn] = zero8();
#pragma unroll
    for (int k0 = 0; k0 < 128; k0 += 32) {
      v16bf a = load_frag_a(&Qs[16 * wave][k0], 128);
#pragma unroll
      for (int tn = 0; tn < 4; tn++) {
        v16bf b = load_frag_b(&Ks[16 * tn][k0], 128);
        sc[tn] = wmma_bf16(a, b, sc[tn]);
      }
    }

    // online softmax in C layout: element (r,lane) = row 8*half+r, col 16*tn+ln
#pragma unroll
    for (int r = 0; r < 8; r++) {
      int rowg = q0 + 16 * wave + 8 * half + r;
      float mx = -1e30f;
#pragma unroll
      for (int tn = 0; tn < 4; tn++) {
        int colg = c0 + 16 * tn + ln;
        float v = sc[tn][r] * scale;
        if (colg > rowg) v = -1e30f;
        sc[tn][r] = v;
        mx = fmaxf(mx, v);
      }
#pragma unroll
      for (int off = 1; off < 16; off <<= 1) mx = fmaxf(mx, __shfl_xor(mx, off, 32));
      float mnew = fmaxf(m_i[r], mx);
      float rsum = 0.f;
#pragma unroll
      for (int tn = 0; tn < 4; tn++) {
        float p = __expf(sc[tn][r] - mnew);
        Ps[wave][8 * half + r][16 * tn + ln] = (bf16)p;
        rsum += p;
      }
#pragma unroll
      for (int off = 1; off < 16; off <<= 1) rsum += __shfl_xor(rsum, off, 32);
      float alpha = __expf(m_i[r] - mnew);
      l_i[r] = l_i[r] * alpha + rsum;
      m_i[r] = mnew;
#pragma unroll
      for (int nt = 0; nt < 8; nt++) acc[nt][r] *= alpha;
    }

    // acc += P (16x64) @ V (64x128); B fragments from transposed V rows
#pragma unroll
    for (int k0 = 0; k0 < 64; k0 += 32) {
      v16bf a = load_frag_a(&Ps[wave][0][k0], 64);
#pragma unroll
      for (int nt = 0; nt < 8; nt++) {
        v16bf b = load_frag_b(&Vt[16 * nt][k0], 64);
        acc[nt] = wmma_bf16(a, b, acc[nt]);
      }
    }
    __syncthreads();
  }

#pragma unroll
  for (int nt = 0; nt < 8; nt++) {
#pragma unroll
    for (int r = 0; r < 8; r++) {
      int rowg = q0 + 16 * wave + 8 * half + r;
      int colg = 16 * nt + ln;
      O[(size_t)rowg * (NHX * HDX) + h * HDX + colg] = (bf16)(acc[nt][r] / l_i[r]);
    }
  }
}

// ------------------------------------------------------------------------ driver
extern "C" void kernel_launch(void* const* d_in, const int* in_sizes, int n_in,
                              void* d_out, int out_size, void* d_ws, size_t ws_size,
                              hipStream_t stream) {
  (void)in_sizes; (void)n_in; (void)out_size; (void)ws_size;
  const float* hs  = (const float*)d_in[0];
  const float* nw  = (const float*)d_in[1];
  const float* qw  = (const float*)d_in[2];
  const float* qb  = (const float*)d_in[3];
  const float* ksw = (const float*)d_in[4];
  const float* ksb = (const float*)d_in[5];
  const float* km0 = (const float*)d_in[6];
  const float* km1 = (const float*)d_in[7];
  const float* kgw = (const float*)d_in[8];
  const float* vsw = (const float*)d_in[9];
  const float* vsb = (const float*)d_in[10];
  const float* vm0 = (const float*)d_in[11];
  const float* vm1 = (const float*)d_in[12];
  const float* vgw = (const float*)d_in[13];
  const float* lrs = (const float*)d_in[14];
  const float* lrm = (const float*)d_in[15];
  const float* dcs = (const float*)d_in[16];
  const float* dcm = (const float*)d_in[17];
  const float* ow  = (const float*)d_in[18];

  char* ws = (char*)d_ws;
  size_t off = 0;
  auto carve = [&](size_t bytes) -> void* {
    void* p = ws + off;
    off += (bytes + 255) & ~(size_t)255;
    return p;
  };
  bf16* xb    = (bf16*)carve((size_t)SEQ * DIMX * 2);
  bf16* qbuf  = (bf16*)carve((size_t)SEQ * DIMX * 2);
  bf16* attn  = (bf16*)carve((size_t)SEQ * DIMX * 2);
  bf16* kbf   = (bf16*)carve((size_t)SEQ * KVD * 2);
  bf16* vbf   = (bf16*)carve((size_t)SEQ * KVD * 2);
  bf16* hbuf  = (bf16*)carve((size_t)SEQ * MEMH * 2);
  float* bufS = (float*)carve((size_t)SEQ * KVD * 4);
  float* bufG = (float*)carve((size_t)SEQ * KVD * 4);
  float* bufM = (float*)carve((size_t)SEQ * KVD * 4);

  float* out    = (float*)d_out;
  float* lr_out = out + (size_t)SEQ * DIMX;
  float* dc_out = lr_out + (size_t)SEQ * NKVX;

  rmsnorm_bf16_kernel<<<SEQ, 256, 0, stream>>>(hs, nw, xb);

  dim3 gq(DIMX / 128, SEQ / 128);   // (28, 16)
  dim3 gkv(KVD / 128, SEQ / 128);   // (4, 16)

  // Q projection (bf16 out for attention)
  gemm_wmma_kernel<EP_BF16_BIAS><<<gq, 256, 0, stream>>>(xb, qw, qb, qbuf, DIMX, DIMX);

  // K adaptive-gated path
  gemm_wmma_kernel<EP_F32_BIAS><<<gkv, 256, 0, stream>>>(xb, ksw, ksb, bufS, DIMX, KVD);
  gemm_wmma_kernel<EP_BF16_SILU><<<gkv, 256, 0, stream>>>(xb, km0, nullptr, hbuf, DIMX, MEMH);
  gemm_wmma_kernel<EP_F32><<<gkv, 256, 0, stream>>>(hbuf, km1, nullptr, bufM, MEMH, KVD);
  gemm_wmma_kernel<EP_F32_SIGMOID><<<gkv, 256, 0, stream>>>(xb, kgw, nullptr, bufG, DIMX, KVD);
  combine_gate_kernel<<<(SEQ * KVD + 255) / 256, 256, 0, stream>>>(bufS, bufG, bufM, kbf, SEQ * KVD);

  // V adaptive-gated path (reuses scratch sequentially)
  gemm_wmma_kernel<EP_F32_BIAS><<<gkv, 256, 0, stream>>>(xb, vsw, vsb, bufS, DIMX, KVD);
  gemm_wmma_kernel<EP_BF16_SILU><<<gkv, 256, 0, stream>>>(xb, vm0, nullptr, hbuf, DIMX, MEMH);
  gemm_wmma_kernel<EP_F32><<<gkv, 256, 0, stream>>>(hbuf, vm1, nullptr, bufM, MEMH, KVD);
  gemm_wmma_kernel<EP_F32_SIGMOID><<<gkv, 256, 0, stream>>>(xb, vgw, nullptr, bufG, DIMX, KVD);
  combine_gate_kernel<<<(SEQ * KVD + 255) / 256, 256, 0, stream>>>(bufS, bufG, bufM, vbf, SEQ * KVD);

  // lr / decay heads
  lrdecay_kernel<<<SEQ, 256, 0, stream>>>(xb, lrs, lrm, dcs, dcm, lr_out, dc_out);

  // causal GQA attention
  flash_wmma_kernel<<<dim3(NHX, SEQ / 64), 128, 0, stream>>>(qbuf, kbf, vbf, attn);

  // output projection (fp32 out)
  gemm_wmma_kernel<EP_F32><<<gq, 256, 0, stream>>>(attn, ow, nullptr, out, DIMX, DIMX);
}
